// DeformableConv_25606595019207
// MI455X (gfx1250) — matrix operand order repaired and checked by reading, not compiled, not run on hardware
//
#include <hip/hip_runtime.h>

// Problem constants (match reference)
#define Bn 8
#define Cc 256
#define Hh 128
#define Ww 128
#define Oo 256
#define Kk 3
#define HW (Hh*Ww)          // 16384
#define KF (Cc*Kk)          // 768 flat-K, ordered kflat = k*Cc + c
#define KCH 32              // K per WMMA step
#define KSTEPS (KF/KCH)     // 24

typedef __bf16 v16bf __attribute__((ext_vector_type(16)));
typedef __bf16 v8bf  __attribute__((ext_vector_type(8)));
typedef float  v8f   __attribute__((ext_vector_type(8)));

__device__ __forceinline__ unsigned short f2bf(float f) {
    unsigned int u = __float_as_uint(f);
    unsigned int r = u + 0x7FFFu + ((u >> 16) & 1u);   // round-to-nearest-even
    return (unsigned short)(r >> 16);
}

// ---------------------------------------------------------------------------
// Phase 0: deform_w (O,C,3) fp32  ->  W2[o][k*C + c] bf16   (A operand, row-major)
// ---------------------------------------------------------------------------
__global__ void prep_weights(const float* __restrict__ dw,
                             unsigned short* __restrict__ W2) {
    int idx = blockIdx.x * 256 + threadIdx.x;      // 0 .. O*KF-1 (196608)
    int o  = idx / KF;
    int kf = idx - o * KF;
    int k  = kf >> 8;                              // kf / 256
    int c  = kf & 255;
    W2[idx] = f2bf(dw[(o * Cc + c) * Kk + k]);
}

// ---------------------------------------------------------------------------
// Phase 1: offset conv (1x3, pad 1) + bias, zero ch1/ch4, emit bilinear meta:
//          meta[b][k][h][w] = {floor(py), floor(px), wy, wx}
// ---------------------------------------------------------------------------
__global__ void offsets_meta(const float* __restrict__ x,
                             const float* __restrict__ ow,
                             const float* __restrict__ ob,
                             float4* __restrict__ meta) {
    __shared__ float wt[6 * 256 * 4];              // [ch][c][4] padded, 24 KB
    int b = blockIdx.x >> 7;
    int h = blockIdx.x & 127;
    int w = threadIdx.x;                           // 128 threads

    for (int i = threadIdx.x; i < 6 * 256 * 4; i += 128) {
        int ch  = i >> 10;
        int rem = i & 1023;
        int c   = rem >> 2;
        int kw  = rem & 3;
        wt[i] = (kw < 3) ? ow[(ch * Cc + c) * 3 + kw] : 0.0f;
    }
    __syncthreads();

    const float* xb = x + ((size_t)(b * Cc) * Hh + h) * Ww;
    float a0 = 0.f, a1 = 0.f, a2 = 0.f, a3 = 0.f, a4 = 0.f, a5 = 0.f;
    const float4* wt4 = (const float4*)wt;

    for (int c = 0; c < Cc; ++c) {
        const float* xr = xb + (size_t)c * HW;
        float xm = (w > 0)      ? xr[w - 1] : 0.0f;
        float xc = xr[w];
        float xp = (w < Ww - 1) ? xr[w + 1] : 0.0f;
        float4 q0 = wt4[0 * 256 + c];
        float4 q1 = wt4[1 * 256 + c];
        float4 q2 = wt4[2 * 256 + c];
        float4 q3 = wt4[3 * 256 + c];
        float4 q4 = wt4[4 * 256 + c];
        float4 q5 = wt4[5 * 256 + c];
        a0 += xm * q0.x + xc * q0.y + xp * q0.z;
        a1 += xm * q1.x + xc * q1.y + xp * q1.z;
        a2 += xm * q2.x + xc * q2.y + xp * q2.z;
        a3 += xm * q3.x + xc * q3.y + xp * q3.z;
        a4 += xm * q4.x + xc * q4.y + xp * q4.z;
        a5 += xm * q5.x + xc * q5.y + xp * q5.z;
    }
    a0 += ob[0]; a2 += ob[2]; a3 += ob[3]; a5 += ob[5];
    a1 = 0.0f;   // offset channel 1 zeroed (k=0 x-offset)
    a4 = 0.0f;   // offset channel 4 zeroed (k=2 y-offset)

    float offy[3] = { a0, a2, a4 };
    float offx[3] = { a1, a3, a5 };
    #pragma unroll
    for (int k = 0; k < Kk; ++k) {
        float py  = (float)h + offy[k];                // base_y = 0 (KH=1)
        float px  = (float)(w + (k - 1)) + offx[k];    // base_x = k-1
        float y0f = floorf(py);
        float x0f = floorf(px);
        meta[(((size_t)b * Kk + k) * Hh + h) * Ww + w] =
            make_float4(y0f, x0f, py - y0f, px - x0f);
    }
}

// ---------------------------------------------------------------------------
// Phase 2: bilinear gather -> bf16 B-tile in LDS -> WMMA GEMM
//   block = one (b,h) row: N = 128 positions, M = 256 out channels, K = 768
//   8 waves: wave wv owns M-tiles {2wv, 2wv+1} x N-tiles {0..7}
// ---------------------------------------------------------------------------
__global__ void __launch_bounds__(256, 1) deform_gemm(
        const float* __restrict__ x,
        const unsigned short* __restrict__ W2,
        const float4* __restrict__ meta,
        float* __restrict__ out) {
    __shared__ __align__(32) unsigned short Bt[128 * KCH];   // 8 KB B tile
    unsigned int* Btu = (unsigned int*)Bt;

    int b    = blockIdx.x >> 7;
    int h    = blockIdx.x & 127;
    int tid  = threadIdx.x;
    int lane = tid & 31;
    int wv   = tid >> 5;          // wave 0..7
    int posT = tid & 127;         // gather position (coalesced: lane ~ w)
    int cIdx = tid >> 7;          // 0/1 -> which 16-channel half
    int g    = lane >> 4;         // lane group 0/1
    int lrow = lane & 15;

    v8f acc[2][8];
    #pragma unroll
    for (int mi = 0; mi < 2; ++mi)
        #pragma unroll
        for (int n = 0; n < 8; ++n) {
            v8f z = {};
            acc[mi][n] = z;
        }

    const float* xb = x + (size_t)b * Cc * HW;
    int  a00 = 0, a01 = 0, a10 = 0, a11 = 0;
    float W00 = 0.f, W01 = 0.f, W10 = 0.f, W11 = 0.f;
    int kcur = -1;

    const unsigned short* Abase0 = W2 + (size_t)((wv * 2 + 0) * 16 + lrow) * KF;
    const unsigned short* Abase1 = W2 + (size_t)((wv * 2 + 1) * 16 + lrow) * KF;

    for (int kc = 0; kc < KSTEPS; ++kc) {
        int k  = kc >> 3;               // tap index (kflat = k*256 + c)
        int c0 = (kc & 7) * KCH;        // channel base of this K chunk
        if (k != kcur) {
            kcur = k;
            float4 m = meta[(((size_t)b * Kk + k) * Hh + h) * Ww + posT];
            int y0 = (int)m.x, x0 = (int)m.y;
            float wy = m.z, wx = m.w;
            int y1 = y0 + 1, x1 = x0 + 1;
            float vy0 = (y0 >= 0 && y0 < Hh) ? 1.f : 0.f;
            float vy1 = (y1 >= 0 && y1 < Hh) ? 1.f : 0.f;
            float vx0 = (x0 >= 0 && x0 < Ww) ? 1.f : 0.f;
            float vx1 = (x1 >= 0 && x1 < Ww) ? 1.f : 0.f;
            int y0c = min(max(y0, 0), Hh - 1), y1c = min(max(y1, 0), Hh - 1);
            int x0c = min(max(x0, 0), Ww - 1), x1c = min(max(x1, 0), Ww - 1);
            a00 = y0c * Ww + x0c;  a01 = y0c * Ww + x1c;
            a10 = y1c * Ww + x0c;  a11 = y1c * Ww + x1c;
            W00 = (1.f - wy) * (1.f - wx) * vy0 * vx0;
            W01 = (1.f - wy) * wx         * vy0 * vx1;
            W10 = wy         * (1.f - wx) * vy1 * vx0;
            W11 = wy         * wx         * vy1 * vx1;
        }

        __syncthreads();   // previous compute done reading Bt
        // Fill B tile: this thread -> position posT, channels c0+cIdx*16 .. +15
        const float* xc = xb + (size_t)(c0 + cIdx * 16) * HW;
        #pragma unroll
        for (int jj = 0; jj < 8; ++jj) {
            const float* p0 = xc + (size_t)(2 * jj) * HW;
            const float* p1 = p0 + HW;
            float v0 = p0[a00]*W00 + p0[a01]*W01 + p0[a10]*W10 + p0[a11]*W11;
            float v1 = p1[a00]*W00 + p1[a01]*W01 + p1[a10]*W10 + p1[a11]*W11;
            Btu[posT * (KCH/2) + cIdx * 8 + jj] =
                (unsigned int)f2bf(v0) | ((unsigned int)f2bf(v1) << 16);
        }
        __syncthreads();   // B tile visible to all waves

        // Prefetch next K-step's weight rows (CDNA5 global_prefetch path);
        // uniform condition -> no EXEC divergence around WMMA below.
        if (kc + 1 < KSTEPS) {
            __builtin_prefetch(Abase0 + (kc + 1) * KCH, 0, 3);
            __builtin_prefetch(Abase1 + (kc + 1) * KCH, 0, 3);
        }

        // A fragments: contiguous K = g*16 + e order (same as B) -> two contiguous 16B runs per lane
        v16bf afr[2];
        {
            union { v16bf v; v8bf hh[2]; } ua;
            const unsigned short* Ar = Abase0 + kc * KCH;
            ua.hh[0] = *(const v8bf*)(Ar + g * 16);       // K = g*16 + e
            ua.hh[1] = *(const v8bf*)(Ar + g * 16 + 8);   // K = g*16 + 8 + e
            afr[0] = ua.v;
            Ar = Abase1 + kc * KCH;
            ua.hh[0] = *(const v8bf*)(Ar + g * 16);
            ua.hh[1] = *(const v8bf*)(Ar + g * 16 + 8);
            afr[1] = ua.v;
        }

        // B fragments from LDS: lane N = lrow, K = g*16 + e (contiguous 32B).
        // One-deep register double-buffer so ds_load of tile n+1 issues before
        // the WMMAs of tile n (hides LDS latency instead of s_wait_dscnt 0).
        v16bf bcur = *(const v16bf*)(Bt + (0 * 16 + lrow) * KCH + g * 16);
        #pragma unroll
        for (int n = 0; n < 8; ++n) {
            v16bf bnxt = bcur;
            if (n < 7)
                bnxt = *(const v16bf*)(Bt + ((n + 1) * 16 + lrow) * KCH + g * 16);
            acc[0][n] = __builtin_amdgcn_wmma_f32_16x16x32_bf16(
                false, afr[0], false, bcur, (short)0, acc[0][n], false, false);
            acc[1][n] = __builtin_amdgcn_wmma_f32_16x16x32_bf16(
                false, afr[1], false, bcur, (short)0, acc[1][n], false, false);
            bcur = bnxt;
        }
    }

    // D layout: lane -> (M = g*8 + r, N = lrow); coalesced f32 stores per row
    #pragma unroll
    for (int mi = 0; mi < 2; ++mi)
        #pragma unroll
        for (int n = 0; n < 8; ++n)
            #pragma unroll
            for (int r = 0; r < 8; ++r) {
                int orow = (wv * 2 + mi) * 16 + g * 8 + r;
                int wcol = n * 16 + lrow;
                out[(((size_t)b * Oo + orow) * Hh + h) * Ww + wcol] = acc[mi][n][r];
            }
}

// ---------------------------------------------------------------------------
extern "C" void kernel_launch(void* const* d_in, const int* in_sizes, int n_in,
                              void* d_out, int out_size, void* d_ws, size_t ws_size,
                              hipStream_t stream) {
    const float* x        = (const float*)d_in[0];
    const float* offset_w = (const float*)d_in[1];
    const float* offset_b = (const float*)d_in[2];
    const float* deform_w = (const float*)d_in[3];
    float* out = (float*)d_out;

    // Workspace: [0, 384KB) bf16 weights, then 6.29MB of float4 gather meta
    unsigned short* W2 = (unsigned short*)d_ws;
    float4* meta = (float4*)((char*)d_ws + (size_t)Oo * KF * sizeof(unsigned short));

    prep_weights<<<(Oo * KF) / 256, 256, 0, stream>>>(deform_w, W2);
    offsets_meta<<<Bn * Hh, 128, 0, stream>>>(x, offset_w, offset_b, meta);
    deform_gemm<<<Bn * Hh, 256, 0, stream>>>(x, W2, meta, out);
}
